// EpistemicQuantizer_86921548137295
// MI455X (gfx1250) — matrix operand order repaired and checked by reading, not compiled, not run on hardware
//
#include <hip/hip_runtime.h>

// ---------------- problem constants ----------------
#define NROWS   16384      // B*T = 8*2048
#define DDIM    512
#define VACT    8192       // ACTIVE_VOCAB
#define M_BLOCK 128        // rows per workgroup (8 waves * 16)
#define CTILE   32         // codebook columns per tile (2 x 16-col subtiles per wave)
#define NTILES  (VACT / CTILE)   // 256
#define KSLICES 16         // 512 / 32

typedef __attribute__((ext_vector_type(16))) __bf16       v16bf;
typedef __attribute__((ext_vector_type(8)))  float        v8f;
typedef __attribute__((ext_vector_type(4)))  unsigned int u32x4;
typedef int vi4 __attribute__((vector_size(16)));          // matches builtin's V4i

#if defined(__has_builtin)
#  if __has_builtin(__builtin_amdgcn_global_load_async_to_lds_b128)
#    define HAVE_ASYNC_LDS 1
#  endif
#  if __has_builtin(__builtin_amdgcn_s_wait_asynccnt)
#    define HAVE_WAIT_ASYNC_BUILTIN 1
#  endif
#endif

union Frag {                  // 32 bytes: one 16-bit WMMA operand per lane
    u32x4 u[2];
    v16bf v;
};

// float -> bf16 (round to nearest even), deterministic
static __device__ __forceinline__ unsigned short f2bf(float f) {
    unsigned int u = __float_as_uint(f);
    u += 0x7FFFu + ((u >> 16) & 1u);
    return (unsigned short)(u >> 16);
}

// ---------------- row L2-normalize -> bf16 ----------------
// one wave (32 lanes) per row, 16 contiguous floats per lane
__global__ __launch_bounds__(256) void normalize_rows(
    const float* __restrict__ in, unsigned short* __restrict__ out, int nrows)
{
    int tid  = threadIdx.x;
    int lane = tid & 31;
    int w    = tid >> 5;
    int row  = blockIdx.x * 8 + w;
    if (row >= nrows) return;

    const float* p = in + (size_t)row * DDIM + lane * 16;
    float4 v[4];
    float s = 0.0f;
#pragma unroll
    for (int i = 0; i < 4; ++i) {
        v[i] = ((const float4*)p)[i];
        s += v[i].x * v[i].x + v[i].y * v[i].y + v[i].z * v[i].z + v[i].w * v[i].w;
    }
#pragma unroll
    for (int off = 1; off < 32; off <<= 1) s += __shfl_xor(s, off, 32);

    float scale = 1.0f / fmaxf(sqrtf(s), 1e-12f);

    unsigned short* o = out + (size_t)row * DDIM + lane * 16;
#pragma unroll
    for (int i = 0; i < 4; ++i) {
        o[i * 4 + 0] = f2bf(v[i].x * scale);
        o[i * 4 + 1] = f2bf(v[i].y * scale);
        o[i * 4 + 2] = f2bf(v[i].z * scale);
        o[i * 4 + 3] = f2bf(v[i].w * scale);
    }
}

// ---------------- zero histogram ----------------
__global__ void zero_counts(int* __restrict__ counts) {
    int i = blockIdx.x * blockDim.x + threadIdx.x;
    if (i < VACT) counts[i] = 0;
}

// ---------------- fused bf16 WMMA GEMM + row argmax ----------------
// block = 256 threads = 8 waves; wave w owns rows [m0 + 16w, +16)
// A (16x512 bf16 per wave) lives in registers; 32-column B tiles double-buffer
// through LDS, staged with gfx1250 async-to-LDS copies when available.
__global__ __launch_bounds__(256) void gemm_argmax(
    const unsigned short* __restrict__ xn,
    const unsigned short* __restrict__ cbn,
    int*   __restrict__ idx_out,
    float* __restrict__ sim_out,
    float* __restrict__ idxf_out,
    int*   __restrict__ counts)
{
    __shared__ unsigned short bt[2][CTILE * DDIM];   // 2 x 32 KB B tiles

    const int tid  = threadIdx.x;
    const int lane = tid & 31;
    const int wave = tid >> 5;
    const int hl   = lane >> 4;     // lane half (0/1)
    const int lm   = lane & 15;
    const int m0   = blockIdx.x * M_BLOCK + wave * 16;

    // ---- load A fragments (16-bit A layout: K groups {0-7,16-23} / {8-15,24-31}) ----
    const unsigned short* aptr = xn + (size_t)(m0 + lm) * DDIM;
    Frag afrag[KSLICES];
#pragma unroll
    for (int ks = 0; ks < KSLICES; ++ks) {
        int kb = ks * 32 + hl * 8;
        afrag[ks].u[0] = *(const u32x4*)(aptr + kb);        // K = kb .. kb+7
        afrag[ks].u[1] = *(const u32x4*)(aptr + kb + 16);   // K = kb+16 .. kb+23
    }

    float bestv[8];
    int   besti[8];
#pragma unroll
    for (int g = 0; g < 8; ++g) { bestv[g] = -3.0e38f; besti[g] = 0; }

    // -------- tile staging: 32 KB per tile, 16 B x 8 per thread --------
#if HAVE_ASYNC_LDS
    // async copy directly into LDS; same INST_OFFSET applies to both addresses
#define STAGE_TILE(T, BUF)                                                          \
    do {                                                                            \
        __attribute__((address_space(1))) vi4* gp =                                 \
            (__attribute__((address_space(1))) vi4*)(unsigned long long)            \
                (cbn + (size_t)(T) * (CTILE * DDIM)) + tid;                         \
        __attribute__((address_space(3))) vi4* lp =                                 \
            (__attribute__((address_space(3))) vi4*)(unsigned int)                  \
            (unsigned long long)(&bt[BUF][0]) + tid;                                \
        __builtin_amdgcn_global_load_async_to_lds_b128(gp, lp, 0,     0);           \
        __builtin_amdgcn_global_load_async_to_lds_b128(gp, lp, 4096,  0);           \
        __builtin_amdgcn_global_load_async_to_lds_b128(gp, lp, 8192,  0);           \
        __builtin_amdgcn_global_load_async_to_lds_b128(gp, lp, 12288, 0);           \
        __builtin_amdgcn_global_load_async_to_lds_b128(gp, lp, 16384, 0);           \
        __builtin_amdgcn_global_load_async_to_lds_b128(gp, lp, 20480, 0);           \
        __builtin_amdgcn_global_load_async_to_lds_b128(gp, lp, 24576, 0);           \
        __builtin_amdgcn_global_load_async_to_lds_b128(gp, lp, 28672, 0);           \
    } while (0)
#  if HAVE_WAIT_ASYNC_BUILTIN
#    define STAGE_WAIT() __builtin_amdgcn_s_wait_asynccnt(0)
#  else
#    define STAGE_WAIT() asm volatile("s_wait_asynccnt 0x0" ::: "memory")
#  endif

    STAGE_TILE(0, 0);
    STAGE_WAIT();
    __syncthreads();
#else
    // fallback: register-staged double buffer
    u32x4 streg[8];
#define LOADT(T)                                                                    \
    do {                                                                            \
        const u32x4* gs = (const u32x4*)(cbn + (size_t)(T) * (CTILE * DDIM)) + tid; \
        _Pragma("unroll") for (int i = 0; i < 8; ++i) streg[i] = gs[i * 256];       \
    } while (0)
#define STORET(BUF)                                                                 \
    do {                                                                            \
        u32x4* ld = (u32x4*)&bt[BUF][0] + tid;                                      \
        _Pragma("unroll") for (int i = 0; i < 8; ++i) ld[i * 256] = streg[i];       \
    } while (0)

    LOADT(0);
    STORET(0);
    __syncthreads();
#endif

    for (int t = 0; t < NTILES; ++t) {
        const int buf = t & 1;
        const int tn  = (t + 1 < NTILES) ? (t + 1) : 0;   // clamped prefetch target

#if HAVE_ASYNC_LDS
        STAGE_TILE(tn, buf ^ 1);          // overlaps with the 32 WMMAs below
#else
        LOADT(tn);                        // global loads in flight during compute
#endif

        v8f acc0 = {0.f, 0.f, 0.f, 0.f, 0.f, 0.f, 0.f, 0.f};
        v8f acc1 = {0.f, 0.f, 0.f, 0.f, 0.f, 0.f, 0.f, 0.f};
#pragma unroll
        for (int ks = 0; ks < KSLICES; ++ks) {
            // B (32x16): lane = column n, half selects K groups {0-7,16-23} / {8-15,24-31} of the slice
            const unsigned short* bp0 = &bt[buf][lm * DDIM + ks * 32 + hl * 8];
            const unsigned short* bp1 = bp0 + 16 * DDIM;
            Frag b0, b1;
            b0.u[0] = *(const u32x4*)bp0;       // ds loads
            b0.u[1] = *(const u32x4*)(bp0 + 16);
            b1.u[0] = *(const u32x4*)bp1;
            b1.u[1] = *(const u32x4*)(bp1 + 16);
            acc0 = __builtin_amdgcn_wmma_f32_16x16x32_bf16(
                false, afrag[ks].v, false, b0.v, (short)0, acc0, false, false);
            acc1 = __builtin_amdgcn_wmma_f32_16x16x32_bf16(
                false, afrag[ks].v, false, b1.v, (short)0, acc1, false, false);
        }

        const int c0 = t * CTILE;
#pragma unroll
        for (int g = 0; g < 8; ++g) {
            // reg g: row m0 + hl*8 + g; subtile 0 col c0+lm, subtile 1 col c0+16+lm
            float v0 = acc0[g];
            if (v0 > bestv[g]) { bestv[g] = v0; besti[g] = c0 + lm; }
            float v1 = acc1[g];
            if (v1 > bestv[g]) { bestv[g] = v1; besti[g] = c0 + 16 + lm; }
        }

#if HAVE_ASYNC_LDS
        STAGE_WAIT();                     // our async writes to bt[buf^1] done
        __syncthreads();                  // everyone done reading bt[buf]
#else
        STORET(buf ^ 1);                  // safe: that buffer's readers barriered at t-1
        __syncthreads();
#endif
    }

    // reduce across the 16 lanes of each half (columns), lowest index wins ties
#pragma unroll
    for (int g = 0; g < 8; ++g) {
        float bv = bestv[g];
        int   bi = besti[g];
#pragma unroll
        for (int off = 1; off < 16; off <<= 1) {
            float ov = __shfl_xor(bv, off, 32);
            int   oi = __shfl_xor(bi, off, 32);
            if (ov > bv || (ov == bv && oi < bi)) { bv = ov; bi = oi; }
        }
        if (lm == 0) {
            int row = m0 + hl * 8 + g;
            idx_out[row]  = bi;
            sim_out[row]  = bv;
            idxf_out[row] = (float)bi;
            atomicAdd(&counts[bi], 1);
        }
    }
}

// ---------------- z_q gather (forward STE value == codebook[idx]) ----------------
__global__ __launch_bounds__(256) void gather_zq(
    const float* __restrict__ cb, const int* __restrict__ idx, float* __restrict__ out)
{
    size_t i = (size_t)blockIdx.x * 256 + threadIdx.x;    // one float4 each
    size_t e = i * 4;
    int n = (int)(e >> 9);          // / 512
    int d = (int)(e & 511);
    float4 v = *(const float4*)(cb + (size_t)idx[n] * DDIM + d);
    *(float4*)(out + e) = v;
}

// ---------------- entropy + mean max-sim (single block, fixed order) ----------------
__global__ __launch_bounds__(256) void finalize_stats(
    const int* __restrict__ counts, const float* __restrict__ sims,
    float* __restrict__ out_ent, float* __restrict__ out_mean)
{
    __shared__ float red[256];
    const int tid = threadIdx.x;
    const float Ninv = (float)NROWS + 1e-9f;

    float ent = 0.0f;
    for (int i = tid; i < VACT; i += 256) {
        float c = (float)counts[i];
        if (c > 0.0f) {
            float p = c / Ninv;
            ent -= p * logf(fmaxf(p, 1e-12f));
        }
    }
    red[tid] = ent;
    __syncthreads();
    for (int s = 128; s > 0; s >>= 1) {
        if (tid < s) red[tid] += red[tid + s];
        __syncthreads();
    }
    if (tid == 0) *out_ent = red[0];
    __syncthreads();

    float sm = 0.0f;
    for (int i = tid; i < NROWS; i += 256) sm += sims[i];
    red[tid] = sm;
    __syncthreads();
    for (int s = 128; s > 0; s >>= 1) {
        if (tid < s) red[tid] += red[tid + s];
        __syncthreads();
    }
    if (tid == 0) *out_mean = red[0] / (float)NROWS;
}

// ---------------- launch ----------------
extern "C" void kernel_launch(void* const* d_in, const int* in_sizes, int n_in,
                              void* d_out, int out_size, void* d_ws, size_t ws_size,
                              hipStream_t stream) {
    (void)in_sizes; (void)n_in; (void)out_size; (void)ws_size;

    const float* x  = (const float*)d_in[0];   // (8,2048,512) f32
    const float* cb = (const float*)d_in[1];   // (16384,512)  f32
    float* out = (float*)d_out;

    // workspace carve (all 16B aligned)
    char* ws = (char*)d_ws;
    unsigned short* xn    = (unsigned short*)(ws);                         // 16 MB
    unsigned short* cbn   = (unsigned short*)(ws + 16777216);              //  8 MB
    int*            idx   = (int*)  (ws + 16777216 + 8388608);             // 64 KB
    float*          sims  = (float*)(ws + 16777216 + 8388608 + 65536);     // 64 KB
    int*            cnts  = (int*)  (ws + 16777216 + 8388608 + 131072);    // 32 KB

    float* zq   = out;                               // N*D
    float* idxf = out + (size_t)NROWS * DDIM;        // N
    float* ent  = idxf + NROWS;                      // 1
    float* mean = ent + 1;                           // 1

    normalize_rows<<<NROWS / 8, 256, 0, stream>>>(x,  xn,  NROWS);
    normalize_rows<<<VACT  / 8, 256, 0, stream>>>(cb, cbn, VACT);
    zero_counts   <<<VACT / 256, 256, 0, stream>>>(cnts);
    gemm_argmax   <<<NROWS / M_BLOCK, 256, 0, stream>>>(xn, cbn, idx, sims, idxf, cnts);
    gather_zq     <<<(NROWS * DDIM) / (4 * 256), 256, 0, stream>>>(cb, idx, zq);
    finalize_stats<<<1, 256, 0, stream>>>(cnts, sims, ent, mean);
}
